// ScaledDotProductAttention_72241349919165
// MI455X (gfx1250) — matrix operand order repaired and checked by reading, compiled, not run on hardware
//
#include <hip/hip_runtime.h>
#include <hip/hip_bf16.h>

typedef __attribute__((ext_vector_type(16))) _Float16 v16h;
typedef __attribute__((ext_vector_type(8)))  float    v8f;

#define S_LEN      2048
#define DKV        64
#define ROW_STRIDE 2052   // 2048 + 4 pad floats: breaks LDS bank conflicts

// One block = one (b, h, 16-query tile). 256 threads = 8 waves (wave32).
__global__ __launch_bounds__(256) void attn_wmma_kernel(
    const float* __restrict__ Q, const float* __restrict__ K,
    const float* __restrict__ V, const unsigned char* __restrict__ Mask,
    float* __restrict__ ctx_out, float* __restrict__ attn_out)
{
  extern __shared__ float sm[];                 // 16 x ROW_STRIDE scores (f32)
  float* red = sm + 16 * ROW_STRIDE;            // 4 * 32 * 8 reduce buffer

  const int tid  = threadIdx.x;
  const int lane = tid & 31;
  const int wid  = tid >> 5;                    // 0..7
  const int m    = lane & 15;                   // A row / B col within tile
  const int hi   = lane >> 4;                   // lane half

  const int blk = blockIdx.x;                   // B*H*(S/16) = 4096
  const int qt  = blk & 127;                    // q-tile within head
  const int bh  = blk >> 7;                     // fused (b,h) 0..31
  const int b   = bh >> 4;
  const int q0  = qt * 16;

  const size_t headOff = (size_t)bh * S_LEN;    // rows into Q/K/V for this head

  // ---------------- Phase 1: scores = (Q K^T) / 8 -> LDS ----------------
  const float* Qrow = Q + (headOff + q0 + m) * DKV;
  v16h aq0, aq1;                                // Q A-fragments, hoisted
  #pragma unroll
  for (int e = 0; e < 16; ++e) {
    int kd = 16 * hi + e;
    aq0[e] = (_Float16)Qrow[kd];
    aq1[e] = (_Float16)Qrow[32 + kd];
  }

  for (int kt = wid; kt < S_LEN / 16; kt += 8) {
    const int k0 = kt * 16;
    const float* Krow = K + (headOff + k0 + m) * DKV;
    if (kt + 8 < S_LEN / 16)
      __builtin_prefetch(K + (headOff + k0 + 128 + m) * DKV, 0, 0);
    v16h bk0, bk1;                              // K^T B-fragments
    #pragma unroll
    for (int e = 0; e < 16; ++e) {
      bk0[e] = (_Float16)Krow[16 * hi + e];
      bk1[e] = (_Float16)Krow[32 + 16 * hi + e];
    }
    v8f c = {};
    c = __builtin_amdgcn_wmma_f32_16x16x32_f16(false, aq0, false, bk0,
                                               (short)0, c, false, false);
    c = __builtin_amdgcn_wmma_f32_16x16x32_f16(false, aq1, false, bk1,
                                               (short)0, c, false, false);
    #pragma unroll
    for (int r = 0; r < 8; ++r)                 // D: M = r + 8*hi, N = m
      sm[(r + 8 * hi) * ROW_STRIDE + k0 + m] = c[r] * 0.125f;  // 1/sqrt(64)
  }
  __syncthreads();

  // ------------- Phase 2: mask + softmax, write attention ---------------
  {
    const int row = tid >> 4;                   // 16 rows x 16 threads
    const int j   = tid & 15;
    float4* srow4 = (float4*)(sm + row * ROW_STRIDE);
    const uchar4* mrow =
        (const uchar4*)(Mask + ((size_t)b * S_LEN + q0 + row) * S_LEN);

    float mx = -3.0e38f;
    #pragma unroll 4
    for (int i = 0; i < 32; ++i) {
      int c4 = j + i * 16;
      float4 s = srow4[c4];
      uchar4 mk = mrow[c4];
      s.x = mk.x ? s.x : 1e-9f;                 // faithful: 1e-9, not -inf
      s.y = mk.y ? s.y : 1e-9f;
      s.z = mk.z ? s.z : 1e-9f;
      s.w = mk.w ? s.w : 1e-9f;
      srow4[c4] = s;
      mx = fmaxf(mx, fmaxf(fmaxf(s.x, s.y), fmaxf(s.z, s.w)));
    }
    #pragma unroll
    for (int d = 8; d >= 1; d >>= 1)            // reduce across 16 lanes
      mx = fmaxf(mx, __shfl_xor(mx, d, 32));

    float sum = 0.f;
    #pragma unroll 4
    for (int i = 0; i < 32; ++i) {
      int c4 = j + i * 16;
      float4 s = srow4[c4];
      s.x = __expf(s.x - mx); s.y = __expf(s.y - mx);
      s.z = __expf(s.z - mx); s.w = __expf(s.w - mx);
      srow4[c4] = s;
      sum += (s.x + s.y) + (s.z + s.w);
    }
    #pragma unroll
    for (int d = 8; d >= 1; d >>= 1)
      sum += __shfl_xor(sum, d, 32);
    const float inv = 1.0f / sum;

    float4* arow4 = (float4*)(attn_out + (headOff + q0 + row) * S_LEN);
    #pragma unroll 4
    for (int i = 0; i < 32; ++i) {
      int c4 = j + i * 16;
      float4 s = srow4[c4];
      s.x *= inv; s.y *= inv; s.z *= inv; s.w *= inv;
      srow4[c4] = s;                            // keep normalized P in LDS
      arow4[c4] = s;                            // single HBM write of attention
    }
  }
  __syncthreads();

  // ---------------- Phase 3: context = P @ V ----------------------------
  {
    const int dt   = wid & 3;                   // d-tile 0..3 (DV = 4*16)
    const int kh   = wid >> 2;                  // k-half 0/1
    const int dcol = dt * 16 + m;
    const float* srowA = sm + m * ROW_STRIDE;   // P row for A-fragment

    v8f c = {};
    for (int kt = kh * 32; kt < kh * 32 + 32; ++kt) {
      const int kblk = kt * 32;
      v16h a, bv;
      #pragma unroll
      for (int e = 0; e < 16; ++e) {
        int kd = 16 * hi + e;
        a[e]  = (_Float16)srowA[kblk + kd];
        bv[e] = (_Float16)V[(headOff + kblk + 16 * hi + e) * DKV + dcol];
      }
      c = __builtin_amdgcn_wmma_f32_16x16x32_f16(false, a, false, bv,
                                                 (short)0, c, false, false);
    }
    if (kh == 1) {                              // partial from upper k-half
      #pragma unroll
      for (int r = 0; r < 8; ++r)
        red[(dt * 32 + lane) * 8 + r] = c[r];
    }
    __syncthreads();
    if (kh == 0) {
      #pragma unroll
      for (int r = 0; r < 8; ++r) {
        float val = c[r] + red[(dt * 32 + lane) * 8 + r];
        ctx_out[(headOff + q0 + r + 8 * hi) * DKV + dcol] = val;
      }
    }
  }
}

extern "C" void kernel_launch(void* const* d_in, const int* in_sizes, int n_in,
                              void* d_out, int out_size, void* d_ws, size_t ws_size,
                              hipStream_t stream) {
  const float* q = (const float*)d_in[0];
  const float* k = (const float*)d_in[1];
  const float* v = (const float*)d_in[2];
  const unsigned char* mask = (const unsigned char*)d_in[3];

  const int B = 2, H = 16, S = 2048, DV = 64;
  float* ctx  = (float*)d_out;                        // (B,H,S,DV)
  float* attn = (float*)d_out + (size_t)B * H * S * DV;  // (B,H,S,S)

  const size_t shmem = (16 * ROW_STRIDE + 4 * 32 * 8) * sizeof(float);
  (void)hipFuncSetAttribute((const void*)attn_wmma_kernel,
                            hipFuncAttributeMaxDynamicSharedMemorySize,
                            (int)shmem);

  dim3 grid(B * H * (S / 16));                        // 4096 blocks
  attn_wmma_kernel<<<grid, 256, shmem, stream>>>(q, k, v, mask, ctx, attn);
}
